// customLSTM_38920993636497
// MI455X (gfx1250) — matrix operand order repaired and checked by reading, not compiled, not run on hardware
//
#include <hip/hip_runtime.h>
#include <hip/hip_bf16.h>

typedef __attribute__((ext_vector_type(16))) _Float16 v16h;
typedef __attribute__((ext_vector_type(8)))  float    v8f;

#define TM1   99
#define BATCH 16384
#define NW    4                       // waves per block
#define NT    (NW * 32)               // threads per block
#define ROWS_PER_BLOCK (NW * 16)

// bias LDS offsets (layer order: fc1 fc2 fc3 fc4 fc5 cfc1 d2v i2h h2h q1 q2 q3 p1 p2 p3)
#define BO_FC1   0
#define BO_FC2   20
#define BO_FC3   50
#define BO_FC4   80
#define BO_FC5   110
#define BO_CFC1  130
#define BO_D2V   150
#define BO_I2H   170
#define BO_H2H   250
#define BO_Q1    330
#define BO_Q2    350
#define BO_Q3    355
#define BO_P1    359
#define BO_P2    379
#define BO_P3    384
#define B_TOTAL  387

#define N_TILES  32    // pre-swizzled 32x16 f16 weight fragments

struct KParams {
  const float* x;        // (99, 16384, 12)
  const float* pat;      // (99, 16384, 4, 3)
  const float* w[15];
  const float* b[15];
  float* out;            // q (99,B,4) ++ xp (99,B,3) ++ rot (99,B,12)
};

// ---------- fragment index maps (ISA 7.12.2, wave32) ----------
// A fragment: lane L holds (m = L&15), element j <-> k = (L>>4)*8 + (j&7) + ((j&8)<<1)
// B fragment: lane L holds (n = L&15), element j <-> k = (L>>4)*8 + (j&7) + ((j&8)<<1)
// C fragment: lane L holds (n = L&15), vgpr r <-> row = r + 8*(L>>4)

__device__ __forceinline__ v8f zero8() {
  v8f z = {0.f, 0.f, 0.f, 0.f, 0.f, 0.f, 0.f, 0.f};
  return z;
}

__device__ __forceinline__ v8f wmma16(v16h a, v16h b, v8f c) {
  return __builtin_amdgcn_wmma_f32_16x16x32_f16(false, a, false, b, (short)0, c,
                                                false, false);
}

// B fragment: contiguous 32B per lane in pre-swizzled LDS -> 2x ds_load_b128
__device__ __forceinline__ v16h load_bf(const _Float16* sWF, int tile, int lane) {
  return *reinterpret_cast<const v16h*>(&sWF[tile * 512 + lane * 16]);
}

// A fragment from A-order staging buffer
__device__ __forceinline__ v16h load_af(const _Float16* aF, int lane) {
  return *reinterpret_cast<const v16h*>(&aF[lane * 16]);
}

// Store C tiles (+bias, optional relu) into A-fragment-order staging (branch-free)
__device__ __forceinline__ void store_afrag(_Float16* dstF, v8f c0, v8f c1,
                                            const float* bias, int ncols, bool relu,
                                            int lane) {
  const int n  = lane & 15;
  const int rb = (lane >> 4) * 8;
  const int n1 = n + 16;
  const float b0 = bias[n < ncols ? n : 0];
  const float b1 = bias[n1 < ncols ? n1 : 0];
  // element (row, col) -> dest lane (row&15)|(((col>>3)&1)<<4), slot (col&7)|(((col>>4)&1)<<3)
  const int dl0 = ((n >> 3) & 1) << 4;          // col = n
  const int jj0 = (n & 7);
  const int dl1 = ((n1 >> 3) & 1) << 4;         // col = n+16
  const int jj1 = (n1 & 7) | 8;
#pragma unroll
  for (int r = 0; r < 8; ++r) {
    const int row = rb + r;
    float v0 = c0[r] + b0;
    if (relu) v0 = fmaxf(v0, 0.0f);
    v0 = (n < ncols) ? v0 : 0.0f;
    dstF[((row & 15) | dl0) * 16 + jj0] = (_Float16)v0;
    float v1 = c1[r] + b1;
    if (relu) v1 = fmaxf(v1, 0.0f);
    v1 = (n1 < ncols) ? v1 : 0.0f;
    dstF[((row & 15) | dl1) * 16 + jj1] = (_Float16)v1;
  }
}

__device__ __forceinline__ float sigmoidf_(float x) {
  return 1.0f / (1.0f + __expf(-x));
}

// ---------- fused kernel: pre-net + LSTM scan + heads + qrot ----------
__global__ __launch_bounds__(NT) void lstm_fused(KParams P) {
  __shared__ alignas(32) _Float16 sWF[N_TILES * 512];   // weight B-fragments (32 KB)
  __shared__ float sB[B_TOTAL];
  __shared__ float sCB[20];                             // cfc1_b + d2v_b
  __shared__ float sPB[80];                             // i2h_b + h2h_b
  __shared__ alignas(32) _Float16 sAF[NW][512];         // A-frag staging (main chain)
  __shared__ alignas(32) _Float16 sAFp[NW][512];        // A-frag staging (p-head)
  __shared__ float sPre[NW][16 * 80];                   // gate activations (after sigmoid/tanh)
  __shared__ float sQ3[NW][16 * 4];
  __shared__ float sP3[NW][16 * 3];

  const int tid  = threadIdx.x;
  const int lane = tid & 31;
  const int wv   = tid >> 5;
  const int row0 = blockIdx.x * ROWS_PER_BLOCK + wv * 16;

  // ---- init: swizzle weights into B-fragment order, zero padding baked in ----
  {
    auto initTile = [&](int tile, const float* wsrc, int din, int dout, int n0) {
      for (int q = tid; q < 512; q += NT) {
        const int ln = q >> 4, j = q & 15;
        const int n = (ln & 15) + n0;
        const int k = (ln >> 4) * 8 + (j & 7) + ((j & 8) << 1);
        sWF[tile * 512 + q] = (k < din && n < dout) ? (_Float16)wsrc[k * dout + n]
                                                    : (_Float16)0.0f;
      }
    };
    initTile(0,  P.w[0], 12, 20, 0);  initTile(1,  P.w[0], 12, 20, 16);  // fc1
    initTile(2,  P.w[1], 20, 30, 0);  initTile(3,  P.w[1], 20, 30, 16);  // fc2
    initTile(4,  P.w[2], 30, 30, 0);  initTile(5,  P.w[2], 30, 30, 16);  // fc3
    initTile(6,  P.w[3], 30, 30, 0);  initTile(7,  P.w[3], 30, 30, 16);  // fc4
    initTile(8,  P.w[4], 30, 20, 0);  initTile(9,  P.w[4], 30, 20, 16);  // fc5
    initTile(10, P.w[5], 20, 20, 0);  initTile(11, P.w[5], 20, 20, 16);  // cell_fc1
    initTile(12, P.w[6], 20, 20, 0);  initTile(13, P.w[6], 20, 20, 16);  // det2vec
    initTile(14, P.w[7], 20, 80, 0);  initTile(15, P.w[7], 20, 80, 16);  // i2h
    initTile(16, P.w[7], 20, 80, 32); initTile(17, P.w[7], 20, 80, 48);
    initTile(18, P.w[7], 20, 80, 64);
    initTile(19, P.w[8], 20, 80, 0);  initTile(20, P.w[8], 20, 80, 16);  // h2h
    initTile(21, P.w[8], 20, 80, 32); initTile(22, P.w[8], 20, 80, 48);
    initTile(23, P.w[8], 20, 80, 64);
    initTile(24, P.w[9], 20, 20, 0);  initTile(25, P.w[9], 20, 20, 16);  // h2q1
    initTile(26, P.w[10], 20, 5, 0);                                     // h2q2
    initTile(27, P.w[11], 5, 4, 0);                                      // h2q3
    initTile(28, P.w[12], 20, 20, 0); initTile(29, P.w[12], 20, 20, 16); // h2p1
    initTile(30, P.w[13], 20, 5, 0);                                     // h2p2
    initTile(31, P.w[14], 5, 3, 0);                                      // h2p3

    auto initBias = [&](int boff, const float* bsrc, int dout) {
      for (int n = tid; n < dout; n += NT) sB[boff + n] = bsrc[n];
    };
    initBias(BO_FC1, P.b[0], 20);  initBias(BO_FC2, P.b[1], 30);
    initBias(BO_FC3, P.b[2], 30);  initBias(BO_FC4, P.b[3], 30);
    initBias(BO_FC5, P.b[4], 20);  initBias(BO_CFC1, P.b[5], 20);
    initBias(BO_D2V, P.b[6], 20);  initBias(BO_I2H, P.b[7], 80);
    initBias(BO_H2H, P.b[8], 80);  initBias(BO_Q1, P.b[9], 20);
    initBias(BO_Q2, P.b[10], 5);   initBias(BO_Q3, P.b[11], 4);
    initBias(BO_P1, P.b[12], 20);  initBias(BO_P2, P.b[13], 5);
    initBias(BO_P3, P.b[14], 3);
    for (int n = tid; n < 20; n += NT) sCB[n] = P.b[5][n] + P.b[6][n];
    for (int n = tid; n < 80; n += NT) sPB[n] = P.b[7][n] + P.b[8][n];
  }
  __syncthreads();

  _Float16* aF  = sAF[wv];
  _Float16* aFp = sAFp[wv];
  float*    pre = sPre[wv];
  float*    q3b = sQ3[wv];
  float*    p3b = sP3[wv];

  const int n_   = lane & 15;
  const int rb_  = (lane >> 4) * 8;
  const int khi8 = (lane >> 4) * 8;
  const int m_   = lane & 15;

  // h (A-fragment layout) and c live entirely in registers
  v16h ah;
#pragma unroll
  for (int j = 0; j < 16; ++j) ah[j] = (_Float16)0.0f;
  float creg[16];
#pragma unroll
  for (int j = 0; j < 16; ++j) creg[j] = 0.0f;

  for (int t = 0; t < TM1; ++t) {
    // ---- fc1: A fragment of x_t straight from HBM (branch-free clamped loads) ----
    {
      const float* xrow = P.x + ((size_t)t * BATCH + row0 + m_) * 12;
      // speculative prefetch of next step's row: same base, constant offset
      // (invalid translation past the end is silently dropped)
      __builtin_prefetch(xrow + (size_t)BATCH * 12, 0, 0);
      v16h ax;
#pragma unroll
      for (int j = 0; j < 16; ++j) {
        const int k = khi8 + (j & 7) + ((j & 8) << 1);
        const float xv = xrow[k < 12 ? k : 0];
        ax[j] = (k < 12) ? (_Float16)xv : (_Float16)0.0f;
      }
      v8f c0 = zero8(), c1 = zero8();
      c0 = wmma16(ax, load_bf(sWF, 0, lane), c0);
      c1 = wmma16(ax, load_bf(sWF, 1, lane), c1);
      store_afrag(aF, c0, c1, sB + BO_FC1, 20, true, lane);
    }
    // ---- fc2..fc5 ----
    { v16h a = load_af(aF, lane); v8f c0 = zero8(), c1 = zero8();
      c0 = wmma16(a, load_bf(sWF, 2, lane), c0);
      c1 = wmma16(a, load_bf(sWF, 3, lane), c1);
      store_afrag(aF, c0, c1, sB + BO_FC2, 30, true, lane); }
    { v16h a = load_af(aF, lane); v8f c0 = zero8(), c1 = zero8();
      c0 = wmma16(a, load_bf(sWF, 4, lane), c0);
      c1 = wmma16(a, load_bf(sWF, 5, lane), c1);
      store_afrag(aF, c0, c1, sB + BO_FC3, 30, true, lane); }
    { v16h a = load_af(aF, lane); v8f c0 = zero8(), c1 = zero8();
      c0 = wmma16(a, load_bf(sWF, 6, lane), c0);
      c1 = wmma16(a, load_bf(sWF, 7, lane), c1);
      store_afrag(aF, c0, c1, sB + BO_FC4, 30, true, lane); }
    { v16h a = load_af(aF, lane); v8f c0 = zero8(), c1 = zero8();
      c0 = wmma16(a, load_bf(sWF, 8, lane), c0);
      c1 = wmma16(a, load_bf(sWF, 9, lane), c1);
      store_afrag(aF, c0, c1, sB + BO_FC5, 20, true, lane); }   // f_t
    // ---- xc = relu(h @ W_cfc1 + f_t @ W_d2v + combined bias) ----
    {
      v16h af_ = load_af(aF, lane);
      v8f c0 = zero8(), c1 = zero8();
      c0 = wmma16(ah,  load_bf(sWF, 10, lane), c0);
      c1 = wmma16(ah,  load_bf(sWF, 11, lane), c1);
      c0 = wmma16(af_, load_bf(sWF, 12, lane), c0);
      c1 = wmma16(af_, load_bf(sWF, 13, lane), c1);
      store_afrag(aF, c0, c1, sCB, 20, true, lane);             // xc
    }
    // ---- preact = xc @ W_i2h + h @ W_h2h (+bias); gate nonlinearities ----
    {
      v16h axc = load_af(aF, lane);
#pragma unroll
      for (int nt = 0; nt < 5; ++nt) {
        v8f c = zero8();
        c = wmma16(axc, load_bf(sWF, 14 + nt, lane), c);
        c = wmma16(ah,  load_bf(sWF, 19 + nt, lane), c);
        const int col = nt * 16 + n_;
        const float bias = sPB[col];
#pragma unroll
        for (int r = 0; r < 8; ++r) {
          float v = c[r] + bias;
          v = (col < 60) ? sigmoidf_(v) : tanhf(v);
          pre[(rb_ + r) * 80 + col] = v;
        }
      }
    }
    // ---- c_t = c*f + i*g ; h_t = o*tanh(c_t) ; h kept as A-fragment regs ----
#pragma unroll
    for (int j = 0; j < 16; ++j) {
      const int k = khi8 + (j & 7) + ((j & 8) << 1);
      if (k < 20) {
        const float gi = pre[m_ * 80 + k];
        const float gf = pre[m_ * 80 + k + 20];
        const float go = pre[m_ * 80 + k + 40];
        const float gg = pre[m_ * 80 + k + 60];
        const float cnew = creg[j] * gf + gi * gg;
        creg[j] = cnew;
        ah[j] = (_Float16)(go * tanhf(cnew));
      }
    }
    // ---- q and p heads, interleaved via separate staging buffers ----
    { v8f c0 = zero8(), c1 = zero8();
      c0 = wmma16(ah, load_bf(sWF, 24, lane), c0);
      c1 = wmma16(ah, load_bf(sWF, 25, lane), c1);
      store_afrag(aF, c0, c1, sB + BO_Q1, 20, true, lane); }    // xq1
    { v8f c0 = zero8(), c1 = zero8();
      c0 = wmma16(ah, load_bf(sWF, 28, lane), c0);
      c1 = wmma16(ah, load_bf(sWF, 29, lane), c1);
      store_afrag(aFp, c0, c1, sB + BO_P1, 20, true, lane); }   // xp1
    { v16h a = load_af(aF, lane); v8f c0 = zero8();
      c0 = wmma16(a, load_bf(sWF, 26, lane), c0);
      store_afrag(aF, c0, zero8(), sB + BO_Q2, 5, true, lane); }   // xq2
    { v16h a = load_af(aFp, lane); v8f c0 = zero8();
      c0 = wmma16(a, load_bf(sWF, 30, lane), c0);
      store_afrag(aFp, c0, zero8(), sB + BO_P2, 5, true, lane); }  // xp2
    { v16h a = load_af(aF, lane); v8f c0 = zero8();
      c0 = wmma16(a, load_bf(sWF, 27, lane), c0);
      if (n_ < 4) {
        const float b = sB[BO_Q3 + n_];
#pragma unroll
        for (int r = 0; r < 8; ++r) q3b[(rb_ + r) * 4 + n_] = c0[r] + b;
      }
    }
    { v16h a = load_af(aFp, lane); v8f c0 = zero8();
      c0 = wmma16(a, load_bf(sWF, 31, lane), c0);
      if (n_ < 3) {
        const float b = sB[BO_P3 + n_];
#pragma unroll
        for (int r = 0; r < 8; ++r) p3b[(rb_ + r) * 3 + n_] = c0[r] + b;
      }
    }
    // ---- normalize q, rotate patterns, write outputs (lanes 0..15, 1 row each) ----
    if (lane < 16) {
      const int row = lane;
      const float xq0 = q3b[row * 4 + 0], xq1 = q3b[row * 4 + 1];
      const float xq2 = q3b[row * 4 + 2], xq3 = q3b[row * 4 + 3];
      const float pv0 = p3b[row * 3 + 0], pv1 = p3b[row * 3 + 1];
      const float pv2 = p3b[row * 3 + 2];
      const float rs = rsqrtf(xq0*xq0 + xq1*xq1 + xq2*xq2 + xq3*xq3);
      const float q0 = xq0*rs, q1 = xq1*rs, q2 = xq2*rs, q3 = xq3*rs;
      const size_t gr = (size_t)t * BATCH + (row0 + row);
      float* qout = P.out + gr * 4;
      qout[0] = q0; qout[1] = q1; qout[2] = q2; qout[3] = q3;
      float* pout = P.out + (size_t)TM1 * BATCH * 4 + gr * 3;
      pout[0] = pv0; pout[1] = pv1; pout[2] = pv2;
      const float* patp = P.pat + gr * 12;
      // speculative prefetch of next step's pattern row (same base, const offset)
      __builtin_prefetch(patp + (size_t)BATCH * 12, 0, 0);
      float* rout = P.out + (size_t)TM1 * BATCH * 7 + gr * 12;
#pragma unroll
      for (int pp = 0; pp < 4; ++pp) {
        const float v0 = patp[pp*3+0], v1 = patp[pp*3+1], v2 = patp[pp*3+2];
        const float uvx = q2*v2 - q3*v1;
        const float uvy = q3*v0 - q1*v2;
        const float uvz = q1*v1 - q2*v0;
        const float uuvx = q2*uvz - q3*uvy;
        const float uuvy = q3*uvx - q1*uvz;
        const float uuvz = q1*uvy - q2*uvx;
        rout[pp*3+0] = v0 + 2.0f*(q0*uvx + uuvx) + pv0;
        rout[pp*3+1] = v1 + 2.0f*(q0*uvy + uuvy) + pv1;
        rout[pp*3+2] = v2 + 2.0f*(q0*uvz + uuvz) + pv2;
      }
    }
  }
}

extern "C" void kernel_launch(void* const* d_in, const int* in_sizes, int n_in,
                              void* d_out, int out_size, void* d_ws, size_t ws_size,
                              hipStream_t stream) {
  (void)d_ws; (void)ws_size; (void)out_size;
  KParams P;
  P.x   = (const float*)d_in[0];
  P.pat = (const float*)d_in[1];
  P.out = (float*)d_out;

  // params-dict flattening is harness-dependent; handle the plausible layouts.
  static const int hDIN[15]  = {12,20,30,30,30,20,20,20,20,20,20,5,20,20,5};
  static const int hDOUT[15] = {20,30,30,30,20,20,20,80,80,20,5,4,20,5,3};
  static const int alpha_w[15] = {9,11,13,15,17,3,1,7,5,25,27,29,19,21,23};
  static const int alpha_b[15] = {8,10,12,14,16,2,0,6,4,24,26,28,18,20,22};

  if (n_in == 3) {
    const float* blob = (const float*)d_in[2];
    size_t off = 0;
    for (int L = 0; L < 15; ++L) {
      P.w[L] = blob + off; off += (size_t)hDIN[L] * hDOUT[L];
      P.b[L] = blob + off; off += (size_t)hDOUT[L];
    }
  } else if (in_sizes[2] == 240) {
    for (int L = 0; L < 15; ++L) {
      P.w[L] = (const float*)d_in[2 + 2 * L];
      P.b[L] = (const float*)d_in[3 + 2 * L];
    }
  } else {
    for (int L = 0; L < 15; ++L) {
      P.w[L] = (const float*)d_in[2 + alpha_w[L]];
      P.b[L] = (const float*)d_in[2 + alpha_b[L]];
    }
  }

  dim3 grid(BATCH / ROWS_PER_BLOCK);   // 256 blocks
  dim3 block(NT);                      // 128 threads = 4 waves
  hipLaunchKernelGGL(lstm_fused, grid, block, 0, stream, P);
}
